// NEXT_Simulator_29678224015913
// MI455X (gfx1250) — matrix operand order repaired and checked by reading, not compiled, not run on hardware
//
#include <hip/hip_runtime.h>

typedef __attribute__((ext_vector_type(16))) _Float16 v16h;
typedef __attribute__((ext_vector_type(8)))  _Float16 v8h;
typedef __attribute__((ext_vector_type(8)))  float    v8f;

#define N_PMTS   12
#define N_TICKS  550
#define NDEP     64
#define KELEC    512
#define BATCH    8
#define WAVES_PER_BLOCK 8
#define THREADS  256
#define ITERS    (KELEC / (WAVES_PER_BLOCK * 32))   // = 2: electrons per wave / 32
#define GAUSS_NORM 3.98942280401f   // 1/(0.1*sqrt(2*pi))
#define EXP_M20   2.0611536e-9f     // exp(-20): per-tick ratio-of-ratios

// ---- gfx1250 async global->LDS path (ASYNCcnt), with safe fallback ----
#if defined(__has_builtin)
# if __has_builtin(__builtin_amdgcn_global_load_async_to_lds_b32)
#  define HAVE_ASYNC 1
# endif
#endif
#ifndef HAVE_ASYNC
# define HAVE_ASYNC 0
#endif

#if HAVE_ASYNC
#define ASYNC_B32(gsrc, ldst)                                            \
    __builtin_amdgcn_global_load_async_to_lds_b32(                       \
        (__attribute__((address_space(1))) int*)(void*)(gsrc),           \
        (__attribute__((address_space(3))) int*)(void*)(ldst), 0, 0)
#endif

__device__ __forceinline__ float fast_sigmoid(float t) {
    return __builtin_amdgcn_rcpf(1.0f + __expf(-t));   // v_exp + v_rcp, no IEEE div
}

__global__ void next_zero_out(float* __restrict__ out, int n) {
    int i = blockIdx.x * blockDim.x + threadIdx.x;
    if (i < n) out[i] = 0.0f;
}

__global__ __launch_bounds__(THREADS)
void next_sim_kernel(const float* __restrict__ edep,    // (B,64,4)
                     const float* __restrict__ e_draws, // (B,64)
                     const float* __restrict__ diff,    // (B,64,512,3)
                     const float* __restrict__ unif,    // (B,64,512)
                     const float* __restrict__ W1,      // (2,28)
                     const float* __restrict__ b1,      // (28)
                     const float* __restrict__ W2,      // (28,12)
                     const float* __restrict__ b2,      // (12)
                     const float* __restrict__ dscale,  // (3)
                     const float* __restrict__ prs,     // (12)
                     const float* __restrict__ lifep,   // (1)
                     float* __restrict__ out)           // (B,12,550)
{
    __shared__ float acc[N_PMTS * N_TICKS];                          // 26400 B
    __shared__ __align__(16) _Float16 hbuf[WAVES_PER_BLOCK][32][32]; // 16 KB
    __shared__ __align__(16) float zmbuf[WAVES_PER_BLOCK][32][2];    // (z, mask)
#if HAVE_ASYNC
    __shared__ __align__(16) float dbuf[WAVES_PER_BLOCK][2][32][4];  // (d0,d1,d2,u) dbl-buffered
#endif

    const int tid  = threadIdx.x;
    const int lane = tid & 31;
    const int wave = tid >> 5;
    const int b    = blockIdx.x >> 6;       // blockIdx.x / NDEP
    const int dep  = blockIdx.x & (NDEP - 1);

    for (int i = tid; i < N_PMTS * N_TICKS; i += THREADS) acc[i] = 0.0f;
    __syncthreads();

    // ---- per-deposit scalars (uniform across the block) ----
    const int   di   = b * NDEP + dep;
    const float px   = edep[di * 4 + 0];
    const float py   = edep[di * 4 + 1];
    const float pz   = edep[di * 4 + 2];
    const float Ev   = edep[di * 4 + 3];
    const float nbar = Ev * 1000.0f * 1000.0f / 22.4f;
    const float sig  = sqrtf(0.15f * nbar);
    int n_elec = (int)(sig * e_draws[di] + nbar);   // trunc toward zero == astype(int32)
    n_elec = min(max(n_elec, 0), KELEC);
    const float sqz  = sqrtf(pz);
    const float ds0  = dscale[0] * dscale[0];
    const float ds1  = dscale[1] * dscale[1];
    const float ds2c = dscale[2] * dscale[2];
    const float inv_life = __builtin_amdgcn_rcpf(lifep[0]);

    // ---- B fragment: W2 zero-padded to 32x16 (KxN), f16 ----
    const int ncol = lane & 15;
    const int kb   = (lane < 16) ? 0 : 8;
    v16h bfrag;
#pragma unroll
    for (int i = 0; i < 16; ++i) {
        const int kkk = kb + i + ((i >= 8) ? 8 : 0);
        const float wv = (ncol < N_PMTS && kkk < 28) ? W2[kkk * N_PMTS + ncol] : 0.0f;
        bfrag[i] = (_Float16)wv;
    }
    const float bias2  = (ncol < N_PMTS) ? b2[ncol] : 0.0f;
    const float pscale = (ncol < N_PMTS) ? prs[ncol] * prs[ncol] : 0.0f;

    // ---- layer-1 weights in registers ----
    float w1x[28], w1y[28], bb1[28];
#pragma unroll
    for (int j = 0; j < 28; ++j) { w1x[j] = W1[j]; w1y[j] = W1[28 + j]; bb1[j] = b1[j]; }

    // wave handles 64 contiguous electrons of this deposit
    const long long kidx0 = (long long)di * KELEC + wave * (ITERS * 32);

#if HAVE_ASYNC
    {   // prologue: async-stage iteration 0 (ASYNCcnt += 4)
        const long long ki = kidx0 + lane;
        ASYNC_B32(diff + ki * 3 + 0, &dbuf[wave][0][lane][0]);
        ASYNC_B32(diff + ki * 3 + 1, &dbuf[wave][0][lane][1]);
        ASYNC_B32(diff + ki * 3 + 2, &dbuf[wave][0][lane][2]);
        ASYNC_B32(unif + ki,         &dbuf[wave][0][lane][3]);
    }
#endif

    for (int iter = 0; iter < ITERS; ++iter) {
        const int k = iter * 32 + lane + wave * (ITERS * 32); // electron within deposit
        float d0, d1, d2, uu;

#if HAVE_ASYNC
        if (iter + 1 < ITERS) {
            const long long ki = kidx0 + (iter + 1) * 32 + lane;
            const int nb = (iter + 1) & 1;
            ASYNC_B32(diff + ki * 3 + 0, &dbuf[wave][nb][lane][0]);
            ASYNC_B32(diff + ki * 3 + 1, &dbuf[wave][nb][lane][1]);
            ASYNC_B32(diff + ki * 3 + 2, &dbuf[wave][nb][lane][2]);
            ASYNC_B32(unif + ki,         &dbuf[wave][nb][lane][3]);
            asm volatile("s_wait_asynccnt 0x4" ::: "memory");  // current buffer landed
        } else {
            asm volatile("s_wait_asynccnt 0x0" ::: "memory");
        }
        d0 = dbuf[wave][iter & 1][lane][0];
        d1 = dbuf[wave][iter & 1][lane][1];
        d2 = dbuf[wave][iter & 1][lane][2];
        uu = dbuf[wave][iter & 1][lane][3];
#else
        {
            const long long ki = kidx0 + iter * 32 + lane;
            if (iter + 1 < ITERS)
                __builtin_prefetch(&diff[(ki + 32) * 3], 0, 0);
            d0 = diff[ki * 3 + 0];
            d1 = diff[ki * 3 + 1];
            d2 = diff[ki * 3 + 2];
            uu = unif[ki];
        }
#endif

        // diffusion + survival
        const float ex = fmaf(ds0 * d0, sqz, px);
        const float ey = fmaf(ds1 * d1, sqz, py);
        const float ez = fmaf(ds2c * d2, sqz, pz);
        const float prob = 1.0f - __expf(-ez * inv_life);
        const float mval = ((k < n_elec) && (prob > uu)) ? 1.0f : 0.0f;

        zmbuf[wave][lane][0] = ez;
        zmbuf[wave][lane][1] = mval;

        // ---- layer 1: h = sigmoid(x*W1_0 + y*W1_1 + b1), staged as f16 rows ----
#pragma unroll
        for (int j = 0; j < 28; ++j) {
            const float t = fmaf(ex, w1x[j], fmaf(ey, w1y[j], bb1[j]));
            hbuf[wave][lane][j] = (_Float16)fast_sigmoid(t);
        }
#pragma unroll
        for (int j = 28; j < 32; ++j) hbuf[wave][lane][j] = (_Float16)0.0f;

        // intra-wave LDS handoff: drain DS counter before cross-lane reads
        asm volatile("s_wait_dscnt 0" ::: "memory");

        // ---- layer 2: two v_wmma_f32_16x16x32_f16 (16 electrons each) ----
#pragma unroll
        for (int g = 0; g < 2; ++g) {
            const int mrow = lane & 15;
            const int koff = (lane < 16) ? 0 : 8;
            const _Float16* hp = &hbuf[wave][g * 16 + mrow][koff];
            const v8h lo = *(const v8h*)(hp);
            const v8h hi = *(const v8h*)(hp + 16);
            const v16h afrag = __builtin_shufflevector(
                lo, hi, 0, 1, 2, 3, 4, 5, 6, 7, 8, 9, 10, 11, 12, 13, 14, 15);

            v8f cz = {};
            const v8f dacc = __builtin_amdgcn_wmma_f32_16x16x32_f16(
                false, afrag, false, bfrag, (short)0, cz, false, false);

            // (z, mask) for this lane's 8 electrons: broadcast LDS vector loads
            const int mbase = g * 16 + ((lane >= 16) ? 8 : 0);
            const float4* zp = (const float4*)&zmbuf[wave][mbase][0];
            const float4 q0 = zp[0], q1 = zp[1], q2 = zp[2], q3 = zp[3];
            const float zz[8] = {q0.x, q0.z, q1.x, q1.z, q2.x, q2.z, q3.x, q3.z};
            const float mm[8] = {q0.y, q0.w, q1.y, q1.w, q2.y, q2.w, q3.y, q3.w};

            if (ncol < N_PMTS) {
#pragma unroll
                for (int r = 0; r < 8; ++r) {
                    if (mm[r] != 0.0f) {
                        const float z  = zz[r];
                        const float pm = pscale * fast_sigmoid(dacc[r] + bias2);
                        int t0 = (int)floorf(z) - 3;
                        t0 = min(max(t0, 0), N_TICKS - 7);
                        const float dl = ((float)t0 + 0.5f) - z;
                        float w  = pm * GAUSS_NORM * __expf(-10.0f * dl * dl);
                        float rr = __expf(fmaf(-20.0f, dl, -10.0f));
                        float* ap = &acc[ncol * N_TICKS + t0];
#pragma unroll
                        for (int tt = 0; tt < 7; ++tt) {
                            atomicAdd(&ap[tt], w);   // ds_add_f32
                            w  *= rr;
                            rr *= EXP_M20;
                        }
                    }
                }
            }
        }
    }

    __syncthreads();
    float* ob = out + (long long)b * N_PMTS * N_TICKS;
    for (int i = tid; i < N_PMTS * N_TICKS; i += THREADS)
        atomicAdd(&ob[i], acc[i]);   // global_atomic_add_f32
}

extern "C" void kernel_launch(void* const* d_in, const int* in_sizes, int n_in,
                              void* d_out, int out_size, void* d_ws, size_t ws_size,
                              hipStream_t stream) {
    (void)in_sizes; (void)n_in; (void)d_ws; (void)ws_size;
    const float* edep    = (const float*)d_in[0];
    const float* e_draws = (const float*)d_in[1];
    const float* diff    = (const float*)d_in[2];
    const float* unif    = (const float*)d_in[3];
    const float* W1      = (const float*)d_in[4];
    const float* b1      = (const float*)d_in[5];
    const float* W2      = (const float*)d_in[6];
    const float* b2      = (const float*)d_in[7];
    const float* dscale  = (const float*)d_in[8];
    const float* prs     = (const float*)d_in[9];
    const float* lifep   = (const float*)d_in[10];
    float* out = (float*)d_out;

    next_zero_out<<<(out_size + 255) / 256, 256, 0, stream>>>(out, out_size);
    // one block per deposit: B*NDEP = 512 blocks, 8 waves each
    next_sim_kernel<<<dim3(BATCH * NDEP), dim3(THREADS), 0, stream>>>(
        edep, e_draws, diff, unif, W1, b1, W2, b2, dscale, prs, lifep, out);
}
